// MultiheadAttentionBlock_60507499266543
// MI455X (gfx1250) — matrix operand order repaired and checked by reading, not compiled, not run on hardware
//
#include <hip/hip_runtime.h>
#include <hip/hip_bf16.h>
#include <stdint.h>

// ---------------- Problem constants ----------------
#define BATCH   4
#define SEQ     2048
#define DMODEL  1024
#define NHEAD   16
#define DK      64
#define MROWS   (BATCH * SEQ)       // 8192

#define USE_TDM 1

typedef __bf16 bf16_t;
typedef __attribute__((ext_vector_type(16))) __bf16 v16bf;
typedef __attribute__((ext_vector_type(8)))  float  v8f;
typedef unsigned int u32x4 __attribute__((ext_vector_type(4)));
typedef int i32x8 __attribute__((ext_vector_type(8)));
typedef int i32x4 __attribute__((ext_vector_type(4)));

// ---------------- small helpers ----------------
__device__ __forceinline__ unsigned short f2bf_u16(float f) {
    union { float f; unsigned u; } x; x.f = f;
    unsigned r = x.u + 0x7FFFu + ((x.u >> 16) & 1u);   // RNE
    return (unsigned short)(r >> 16);
}
__device__ __forceinline__ bf16_t f2bf(float f) {
    union { unsigned short s; bf16_t b; } y; y.s = f2bf_u16(f); return y.b;
}
__device__ __forceinline__ v8f vzero8() {
    v8f z;
#pragma unroll
    for (int i = 0; i < 8; ++i) z[i] = 0.0f;
    return z;
}
__device__ __forceinline__ v8f wmma_bf16(v16bf a, v16bf b, v8f c) {
    return __builtin_amdgcn_wmma_f32_16x16x32_bf16(
        false, a, false, b, (short)0, c, false, false);
}

// A fragment: 16x32 bf16, source row-major (K contiguous), ld in elements.
__device__ __forceinline__ v16bf load_frag_a(const bf16_t* base, int ld, int row0, int k0) {
    const int lane = threadIdx.x & 31;
    const int half = lane >> 4;
    const int r    = lane & 15;
    const bf16_t* p = base + (size_t)(row0 + r) * ld + k0 + (half ? 8 : 0);
    union { v16bf v; uint4 q[2]; } u;
    u.q[0] = *reinterpret_cast<const uint4*>(p);
    u.q[1] = *reinterpret_cast<const uint4*>(p + 16);
    return u.v;
}

// B fragment: 32x16 bf16 (KxN); B[k][n] lives at base[(col0+n)*ld + k].
__device__ __forceinline__ v16bf load_frag_b(const bf16_t* base, int ld, int col0, int k0) {
    const int lane = threadIdx.x & 31;
    const int half = lane >> 4;
    const int n    = lane & 15;
    const bf16_t* p = base + (size_t)(col0 + n) * ld + k0 + (half ? 8 : 0);
    union { v16bf v; uint4 q[2]; } u;
    u.q[0] = *reinterpret_cast<const uint4*>(p);
    u.q[1] = *reinterpret_cast<const uint4*>(p + 16);
    return u.v;
}

#if USE_TDM
// Issue a TDM 2D tile load (bf16 elements): global (row-major, stride0 elems)
// -> LDS at byte offset lds_off. Descriptor layout per CDNA5 ISA ch.8.
// clang-23 toolchain: 6-arg builtin (g0, g1, g2, g3, g_extra, cpol).
__device__ __forceinline__ void tdm_load_tile_2d(const void* gptr, unsigned lds_off,
                                                 unsigned tile_d0, unsigned tile_d1,
                                                 unsigned tensor_d0, unsigned tensor_d1,
                                                 unsigned long long stride0_elems) {
    const unsigned long long ga = (unsigned long long)(uintptr_t)gptr;
    u32x4 g0;
    g0[0] = 0x1u;                                           // count=1, user mode
    g0[1] = lds_off;                                        // lds_addr (bytes)
    g0[2] = (unsigned)(ga & 0xFFFFFFFFu);                   // global_addr[31:0]
    g0[3] = (unsigned)((ga >> 32) & 0x01FFFFFFu)            // global_addr[56:32]
          | (2u << 30);                                     // type=2 ("image")
    i32x8 g1;
    g1[0] = (int)(1u << 16);                                // data_size=1 (2B), wg_mask=0
    g1[1] = (int)((tensor_d0 & 0xFFFFu) << 16);             // tensor_dim0[15:0]
    g1[2] = (int)(((tensor_d0 >> 16) & 0xFFFFu) | ((tensor_d1 & 0xFFFFu) << 16));
    g1[3] = (int)(((tensor_d1 >> 16) & 0xFFFFu) | ((tile_d0 & 0xFFFFu) << 16));
    g1[4] = (int)(tile_d1 & 0xFFFFu);                       // tile_dim1; tile_dim2=0
    g1[5] = (int)(stride0_elems & 0xFFFFFFFFull);           // dim0 stride lo
    g1[6] = (int)((stride0_elems >> 32) & 0xFFFFull);       // dim0 stride hi; dim1 stride=0
    g1[7] = 0;
    i32x4 z4; z4[0] = z4[1] = z4[2] = z4[3] = 0;
    i32x8 z8;
#pragma unroll
    for (int i = 0; i < 8; ++i) z8[i] = 0;
    __builtin_amdgcn_tensor_load_to_lds(g0, g1, z4, z4, z8, 0);
}
#endif

// ---------------- fp32 -> bf16 packed converter ----------------
__global__ void cvt_bf16_kernel(const float* __restrict__ src,
                                bf16_t* __restrict__ dst, int n8) {
    for (int i = blockIdx.x * blockDim.x + threadIdx.x; i < n8;
         i += gridDim.x * blockDim.x) {
        const float4 a = reinterpret_cast<const float4*>(src)[2 * i + 0];
        const float4 b = reinterpret_cast<const float4*>(src)[2 * i + 1];
        union { uint4 q; unsigned short s[8]; } u;
        u.s[0] = f2bf_u16(a.x); u.s[1] = f2bf_u16(a.y);
        u.s[2] = f2bf_u16(a.z); u.s[3] = f2bf_u16(a.w);
        u.s[4] = f2bf_u16(b.x); u.s[5] = f2bf_u16(b.y);
        u.s[6] = f2bf_u16(b.z); u.s[7] = f2bf_u16(b.w);
        reinterpret_cast<uint4*>(dst)[i] = u.q;
    }
}

// ---------------- GEMM: out = X @ W^T + bias ----------------
// MODE 0: Q -> bf16 [b,h,s,d]   MODE 1: K -> bf16 [b,h,s,d]
// MODE 2: V -> bf16 [b,h,d,s]   MODE 3: fp32 [m,n]
template <int MODE>
__global__ __launch_bounds__(256, 1)
void gemm_kernel(const bf16_t* __restrict__ X,
                 const bf16_t* __restrict__ W,
                 const float* __restrict__ bias,
                 void* __restrict__ dstv) {
    const int wave = threadIdx.x >> 5;
    const int lane = threadIdx.x & 31;
    const int half = lane >> 4;
    const int n    = lane & 15;
    const int m0 = blockIdx.y * 128 + wave * 16;
    const int n0 = blockIdx.x * 64;
    const bf16_t* Xb = X + (size_t)m0 * DMODEL;

    v8f acc[4];
#pragma unroll
    for (int j = 0; j < 4; ++j) acc[j] = vzero8();

    for (int k0 = 0; k0 < DMODEL; k0 += 32) {
        if (k0 + 32 < DMODEL)
            __builtin_prefetch(Xb + ((lane & 15) * DMODEL) + k0 + 32, 0, 0);
        const v16bf a = load_frag_a(Xb, DMODEL, 0, k0);
#pragma unroll
        for (int j = 0; j < 4; ++j) {
            const v16bf b = load_frag_b(W, DMODEL, n0 + j * 16, k0);
            acc[j] = wmma_bf16(a, b, acc[j]);
        }
    }

#pragma unroll
    for (int j = 0; j < 4; ++j) {
        const int col = n0 + j * 16 + n;
        const float bv = bias[col];
#pragma unroll
        for (int r = 0; r < 8; ++r) {
            const int row = m0 + r + half * 8;
            const float v = acc[j][r] + bv;
            if (MODE == 3) {
                reinterpret_cast<float*>(dstv)[(size_t)row * DMODEL + col] = v;
            } else {
                const int bb = row >> 11, s = row & (SEQ - 1);
                const int hd = col >> 6,  d = col & (DK - 1);
                bf16_t* dst = reinterpret_cast<bf16_t*>(dstv);
                if (MODE == 2)
                    dst[(((size_t)bb * NHEAD + hd) * DK + d) * SEQ + s] = f2bf(v);
                else
                    dst[(((size_t)bb * NHEAD + hd) * SEQ + s) * DK + d] = f2bf(v);
            }
        }
    }
}

// ---------------- Flash attention (causal) ----------------
__global__ __launch_bounds__(128, 1)
void attn_kernel(const bf16_t* __restrict__ Qh,
                 const bf16_t* __restrict__ Kh,
                 const bf16_t* __restrict__ Vt,
                 bf16_t* __restrict__ ctx) {
    const int bh   = blockIdx.y;
    const int b    = bh >> 4;
    const int h    = bh & (NHEAD - 1);
    const int q0   = blockIdx.x * 64;
    const int wave = threadIdx.x >> 5;
    const int lane = threadIdx.x & 31;
    const int half = lane >> 4;
    const int n    = lane & 15;

    __shared__ __align__(16) bf16_t KtS[64 * 64];        // [kk][d]
    __shared__ __align__(16) bf16_t VtS[64 * 64];        // [d][kk]
    __shared__ __align__(16) bf16_t PS[4][16 * 64];      // per-wave P tile

    const bf16_t* Qbase = Qh + (((size_t)bh * SEQ) + q0 + wave * 16) * DK;
    const v16bf aq0 = load_frag_a(Qbase, DK, 0, 0);
    const v16bf aq1 = load_frag_a(Qbase, DK, 0, 32);

    v8f acc[4];
#pragma unroll
    for (int j = 0; j < 4; ++j) acc[j] = vzero8();
    float mrow[8], lrow[8];
#pragma unroll
    for (int r = 0; r < 8; ++r) { mrow[r] = -INFINITY; lrow[r] = 0.0f; }

    const bf16_t* Kbase = Kh + (size_t)bh * SEQ * DK;
    const bf16_t* Vbase = Vt + (size_t)bh * DK * SEQ;
    const int kend = q0 + 64;

    for (int kk0 = 0; kk0 < kend; kk0 += 64) {
        __syncthreads();                      // previous tile fully consumed
#if USE_TDM
        // Wave 0 drives the Tensor Data Mover for both tiles, waits TENSORcnt.
        if (wave == 0) {
            tdm_load_tile_2d(Kbase + (size_t)kk0 * DK,
                             (unsigned)(uintptr_t)(void*)KtS,
                             /*tile*/ DK, 64, /*tensor*/ DK, 64,
                             /*stride0*/ DK);
            tdm_load_tile_2d(Vbase + kk0,
                             (unsigned)(uintptr_t)(void*)VtS,
                             /*tile*/ 64, DK, /*tensor*/ 64, DK,
                             /*stride0*/ SEQ);
            __builtin_amdgcn_s_wait_tensorcnt(0);
        }
#else
        for (int i = threadIdx.x; i < 512; i += 128) {
            const int row = i >> 3, cc = i & 7;
            reinterpret_cast<uint4*>(KtS)[i] =
                *reinterpret_cast<const uint4*>(Kbase + (size_t)(kk0 + row) * DK + cc * 8);
            reinterpret_cast<uint4*>(VtS)[i] =
                *reinterpret_cast<const uint4*>(Vbase + (size_t)row * SEQ + kk0 + cc * 8);
        }
#endif
        __syncthreads();

        // Scores: S(16x64) = Q(16x64) @ K^T, scaled + causal mask.
        float sc[4][8];
#pragma unroll
        for (int c = 0; c < 4; ++c) {
            v8f s = vzero8();
            s = wmma_bf16(aq0, load_frag_b(KtS, DK, c * 16, 0),  s);
            s = wmma_bf16(aq1, load_frag_b(KtS, DK, c * 16, 32), s);
#pragma unroll
            for (int r = 0; r < 8; ++r) {
                const int col = kk0 + c * 16 + n;
                const int row = q0 + wave * 16 + r + half * 8;
                const float v = s[r] * 0.125f;           // 1/sqrt(64)
                sc[c][r] = (col <= row) ? v : -INFINITY;
            }
        }

        // Online softmax per row (rows live in 16-lane halves).
#pragma unroll
        for (int r = 0; r < 8; ++r) {
            float tm = fmaxf(fmaxf(sc[0][r], sc[1][r]), fmaxf(sc[2][r], sc[3][r]));
            tm = fmaxf(tm, __shfl_xor(tm, 1, 32));
            tm = fmaxf(tm, __shfl_xor(tm, 2, 32));
            tm = fmaxf(tm, __shfl_xor(tm, 4, 32));
            tm = fmaxf(tm, __shfl_xor(tm, 8, 32));
            const float mnew  = fmaxf(mrow[r], tm);
            const float alpha = __expf(mrow[r] - mnew);
            float rsum = 0.0f;
#pragma unroll
            for (int c = 0; c < 4; ++c) {
                const float p = __expf(sc[c][r] - mnew);
                sc[c][r] = p;
                rsum += p;
            }
            rsum += __shfl_xor(rsum, 1, 32);
            rsum += __shfl_xor(rsum, 2, 32);
            rsum += __shfl_xor(rsum, 4, 32);
            rsum += __shfl_xor(rsum, 8, 32);
            lrow[r] = lrow[r] * alpha + rsum;
            mrow[r] = mnew;
#pragma unroll
            for (int j = 0; j < 4; ++j) acc[j][r] *= alpha;
        }

        // C-layout P -> LDS (bf16) -> A-layout fragments (same-wave, DS in-order)
#pragma unroll
        for (int c = 0; c < 4; ++c)
#pragma unroll
            for (int r = 0; r < 8; ++r)
                PS[wave][(r + half * 8) * 64 + c * 16 + n] = f2bf(sc[c][r]);

        const v16bf ap0 = load_frag_a(&PS[wave][0], 64, 0, 0);
        const v16bf ap1 = load_frag_a(&PS[wave][0], 64, 0, 32);
#pragma unroll
        for (int j = 0; j < 4; ++j) {
            acc[j] = wmma_bf16(ap0, load_frag_b(VtS, DK, j * 16, 0),  acc[j]);
            acc[j] = wmma_bf16(ap1, load_frag_b(VtS, DK, j * 16, 32), acc[j]);
        }
    }

    // Normalize, store context bf16 [b, s, h*64+d] (A-ready for the Wo GEMM).
#pragma unroll
    for (int j = 0; j < 4; ++j)
#pragma unroll
        for (int r = 0; r < 8; ++r) {
            const int row = q0 + wave * 16 + r + half * 8;
            const int col = h * DK + j * 16 + n;
            ctx[((size_t)b * SEQ + row) * DMODEL + col] = f2bf(acc[j][r] / lrow[r]);
        }
}

// ---------------- host-side launch ----------------
extern "C" void kernel_launch(void* const* d_in, const int* in_sizes, int n_in,
                              void* d_out, int out_size, void* d_ws, size_t ws_size,
                              hipStream_t stream) {
    (void)in_sizes; (void)n_in; (void)out_size;
    const size_t MB = 1ull << 20;
    if (ws_size < 120 * MB) return;

    char* ws = reinterpret_cast<char*>(d_ws);
    bf16_t* qbf = (bf16_t*)(ws + 0   * MB);
    bf16_t* kbf = (bf16_t*)(ws + 16  * MB);
    bf16_t* vbf = (bf16_t*)(ws + 32  * MB);
    bf16_t* wqb = (bf16_t*)(ws + 48  * MB);
    bf16_t* wkb = (bf16_t*)(ws + 50  * MB);
    bf16_t* wvb = (bf16_t*)(ws + 52  * MB);
    bf16_t* wob = (bf16_t*)(ws + 54  * MB);
    bf16_t* Qh  = (bf16_t*)(ws + 56  * MB);
    bf16_t* Kh  = (bf16_t*)(ws + 72  * MB);
    bf16_t* Vt  = (bf16_t*)(ws + 88  * MB);
    bf16_t* ctx = (bf16_t*)(ws + 104 * MB);

    const float* q  = (const float*)d_in[0];
    const float* k  = (const float*)d_in[1];
    const float* v  = (const float*)d_in[2];
    const float* Wq = (const float*)d_in[4];  const float* bq = (const float*)d_in[5];
    const float* Wk = (const float*)d_in[6];  const float* bk = (const float*)d_in[7];
    const float* Wv = (const float*)d_in[8];  const float* bv = (const float*)d_in[9];
    const float* Wo = (const float*)d_in[10]; const float* bo = (const float*)d_in[11];
    float* out = (float*)d_out;

    const int nAct8 = (MROWS * DMODEL) / 8;
    const int nW8   = (DMODEL * DMODEL) / 8;
    cvt_bf16_kernel<<<2048, 256, 0, stream>>>(q,  qbf, nAct8);
    cvt_bf16_kernel<<<2048, 256, 0, stream>>>(k,  kbf, nAct8);
    cvt_bf16_kernel<<<2048, 256, 0, stream>>>(v,  vbf, nAct8);
    cvt_bf16_kernel<<<512,  256, 0, stream>>>(Wq, wqb, nW8);
    cvt_bf16_kernel<<<512,  256, 0, stream>>>(Wk, wkb, nW8);
    cvt_bf16_kernel<<<512,  256, 0, stream>>>(Wv, wvb, nW8);
    cvt_bf16_kernel<<<512,  256, 0, stream>>>(Wo, wob, nW8);

    dim3 gG(DMODEL / 64, MROWS / 128);
    gemm_kernel<0><<<gG, 256, 0, stream>>>(qbf, wqb, bq, (void*)Qh);
    gemm_kernel<1><<<gG, 256, 0, stream>>>(kbf, wkb, bk, (void*)Kh);
    gemm_kernel<2><<<gG, 256, 0, stream>>>(vbf, wvb, bv, (void*)Vt);

    dim3 gA(SEQ / 64, BATCH * NHEAD);
    attn_kernel<<<gA, 128, 0, stream>>>(Qh, Kh, Vt, ctx);

    gemm_kernel<3><<<gG, 256, 0, stream>>>(ctx, wob, bo, (void*)out);
}
